// CustomInt8Linear_66769561584242
// MI455X (gfx1250) — matrix operand order repaired and checked by reading, not compiled, not run on hardware
//
#include <hip/hip_runtime.h>

typedef __attribute__((ext_vector_type(16))) _Float16 v16h;
typedef __attribute__((ext_vector_type(8)))  _Float16 v8h;
typedef __attribute__((ext_vector_type(4)))  _Float16 v4h;
typedef __attribute__((ext_vector_type(8)))  float    v8f;

#define M_DIM 4096   // B*S = 8*512
#define N_DIM 4096   // OUT_F
#define K_DIM 4096   // IN_F
#define LDSS 72      // f16 elems per LDS row: 64 + 8 pad -> 144B stride (16B aligned, conflict-free frags)

union Frag { uint4 u[2]; v16h h; };

// A operand fragment (16x32, MxK). ISA 16-bit A layout: lane%16 = M row,
// K chunks [kb, kb+8) and [kb+16, kb+24) with kb = (lane<16 ? 0 : 8) + kk.
__device__ __forceinline__ v16h frag_a(const _Float16* s, int row_base, int lane, int kk) {
    const int r  = row_base + (lane & 15);
    const int kb = kk + ((lane >> 4) << 3);
    Frag f;
    f.u[0] = *(const uint4*)(s + r * LDSS + kb);
    f.u[1] = *(const uint4*)(s + r * LDSS + kb + 16);
    return f.h;
}

// B operand fragment (32x16, KxN), stored row-major-along-K (W is [N,K]).
// ISA 16-bit B layout: lane%16 = N col, K chunks [kb, kb+8) and
// [kb+16, kb+24) with kb = (lane<16 ? 0 : 8) + kk (same as A).
__device__ __forceinline__ v16h frag_b(const _Float16* s, int row_base, int lane, int kk) {
    const int r  = row_base + (lane & 15);
    const int kb = kk + ((lane >> 4) << 3);
    Frag f;
    f.u[0] = *(const uint4*)(s + r * LDSS + kb);
    f.u[1] = *(const uint4*)(s + r * LDSS + kb + 16);
    return f.h;
}

// CDNA5 async global->LDS DMA: 16B per lane, tracked by ASYNCcnt (no VGPR transit).
__device__ __forceinline__ void async_b128(unsigned lds_byte_off, const void* sbase, unsigned voff_bytes) {
    asm volatile("global_load_async_to_lds_b128 %0, %1, %2"
                 :: "v"(lds_byte_off), "v"(voff_bytes), "s"(sbase)
                 : "memory");
}
__device__ __forceinline__ void wait_asynccnt0() {
    asm volatile("s_wait_asynccnt 0" ::: "memory");
}

// ---------------- pass 1: one-shot dtype conversion into workspace ----------------

__global__ __launch_bounds__(256) void convert_a_kernel(const float* __restrict__ A,
                                                        _Float16* __restrict__ Ah) {
    const size_t i = ((size_t)blockIdx.x * 256 + threadIdx.x) * 8;
    const float4 a = *(const float4*)(A + i);
    const float4 b = *(const float4*)(A + i + 4);
    v8h h;
    h[0] = (_Float16)a.x; h[1] = (_Float16)a.y; h[2] = (_Float16)a.z; h[3] = (_Float16)a.w;
    h[4] = (_Float16)b.x; h[5] = (_Float16)b.y; h[6] = (_Float16)b.z; h[7] = (_Float16)b.w;
    *(v8h*)(Ah + i) = h;
}

__global__ __launch_bounds__(256) void convert_w_kernel(const int* __restrict__ W,
                                                        _Float16* __restrict__ Wh) {
    const size_t i = ((size_t)blockIdx.x * 256 + threadIdx.x) * 8;
    const int4 a = *(const int4*)(W + i);
    const int4 b = *(const int4*)(W + i + 4);
    v8h h;  // int8-range values: exact in f16; short-cast enables v_cvt_f16_i16
    h[0] = (_Float16)(short)a.x; h[1] = (_Float16)(short)a.y;
    h[2] = (_Float16)(short)a.z; h[3] = (_Float16)(short)a.w;
    h[4] = (_Float16)(short)b.x; h[5] = (_Float16)(short)b.y;
    h[6] = (_Float16)(short)b.z; h[7] = (_Float16)(short)b.w;
    *(v8h*)(Wh + i) = h;
}

// ---------------- pass 2: f16 GEMM, 128x256 block, 64x64 wave tiles, ----------------
// ---------------- async-DMA double-buffered LDS (no register staging)  ----------------

#define GBM 128
#define GBN 256
#define GBK 64
#define ABUF (GBM * LDSS)   // halves per A buffer
#define BBUF (GBN * LDSS)   // halves per B buffer

__global__ __launch_bounds__(256)
void gemm_f16_wmma_kernel(const _Float16* __restrict__ Ah,  // [M, K]
                          const _Float16* __restrict__ Wh,  // [N, K]
                          const float* __restrict__ scale,
                          const float* __restrict__ bias,
                          float* __restrict__ Out)          // [M, N]
{
    __shared__ _Float16 sA[2 * ABUF];
    __shared__ _Float16 sB[2 * BBUF];

    const int tid  = threadIdx.x;
    const int lane = tid & 31;
    const int wave = tid >> 5;

    const long block_m = (long)blockIdx.y * GBM;
    const long block_n = (long)blockIdx.x * GBN;

    // 8 waves: 2 (M) x 4 (N), each owns a 64x64 tile -> 4x4 WMMA tiles
    const int wm = (wave & 1) * 64;
    const int wn = (wave >> 1) * 64;

    v8f acc[4][4] = {};

    // per-thread DMA mapping: rows of 64 f16 = 8 chunks of 16B; thread -> (row0+32j, cH)
    const int row0 = tid >> 3;        // 0..31
    const int cH   = (tid & 7) * 8;   // half-element offset, 16B aligned

    // tile-local SGPR bases for GVS addressing
    const void* baseA = (const void*)(Ah + (size_t)block_m * K_DIM);
    const void* baseB = (const void*)(Wh + (size_t)block_n * K_DIM);

    // LDS byte addresses (generic-pointer low 32 bits == LDS byte offset)
    const unsigned sAb = (unsigned)(size_t)&sA[0];
    const unsigned sBb = (unsigned)(size_t)&sB[0];

    // issue the async fill of buffer `p` with k-block starting at k0
    auto stage = [&](int k0, int p) {
        const unsigned la0 = sAb + (unsigned)(p * ABUF + row0 * LDSS + cH) * 2u;
        const unsigned lb0 = sBb + (unsigned)(p * BBUF + row0 * LDSS + cH) * 2u;
        const unsigned ga0 = (unsigned)((row0 * K_DIM + k0 + cH) * 2);
        #pragma unroll
        for (int j = 0; j < 4; ++j)
            async_b128(la0 + (unsigned)(j * 32 * LDSS) * 2u, baseA,
                       ga0 + (unsigned)(j * 32 * K_DIM) * 2u);
        #pragma unroll
        for (int j = 0; j < 8; ++j)
            async_b128(lb0 + (unsigned)(j * 32 * LDSS) * 2u, baseB,
                       ga0 + (unsigned)(j * 32 * K_DIM) * 2u);
    };

    stage(0, 0);
    wait_asynccnt0();
    __syncthreads();

    int p = 0;
    for (int k0 = 0; k0 < K_DIM; k0 += GBK, p ^= 1) {
        const int k1 = k0 + GBK;
        if (k1 < K_DIM) {
            stage(k1, p ^ 1);   // DMA next tile while computing on this one
            __builtin_prefetch(Ah + (size_t)(block_m + row0) * K_DIM + k1 + GBK + cH, 0, 1);
            __builtin_prefetch(Wh + (size_t)(block_n + row0) * K_DIM + k1 + GBK + cH, 0, 1);
        }

        const _Float16* cA = sA + p * ABUF;
        const _Float16* cB = sB + p * BBUF;

        #pragma unroll
        for (int kk = 0; kk < GBK; kk += 32) {
            v16h af[4], bf[4];
            #pragma unroll
            for (int mi = 0; mi < 4; ++mi) af[mi] = frag_a(cA, wm + mi * 16, lane, kk);
            #pragma unroll
            for (int ni = 0; ni < 4; ++ni) bf[ni] = frag_b(cB, wn + ni * 16, lane, kk);
            #pragma unroll
            for (int mi = 0; mi < 4; ++mi)
                #pragma unroll
                for (int ni = 0; ni < 4; ++ni)
                    acc[mi][ni] = __builtin_amdgcn_wmma_f32_16x16x32_f16(
                        false, af[mi], false, bf[ni], (short)0, acc[mi][ni], false, false);
        }

        if (k1 < K_DIM) {
            wait_asynccnt0();   // our DMA done; barrier makes everyone's visible
            __syncthreads();    // also protects WAR on buffer p (refilled next iter)
        }
    }

    const float s = scale[0];
    #pragma unroll
    for (int mi = 0; mi < 4; ++mi) {
        const long m0 = block_m + wm + mi * 16 + ((lane >> 4) << 3);
        #pragma unroll
        for (int ni = 0; ni < 4; ++ni) {
            const long n   = block_n + wn + ni * 16 + (lane & 15);
            const float bv = bias[n];
            #pragma unroll
            for (int r = 0; r < 8; ++r)
                Out[(m0 + r) * (long)N_DIM + n] = acc[mi][ni][r] * s + bv;
        }
    }
}

// ---------------- fallback: fused convert+GEMM (if workspace too small) ----------------

#define FBM 128
#define FBN 128
#define FBK 64

__global__ __launch_bounds__(256)
void fused_int8_linear_kernel(const float* __restrict__ A, const int* __restrict__ Wq,
                              const float* __restrict__ scale, const float* __restrict__ bias,
                              float* __restrict__ Out)
{
    __shared__ _Float16 sA[FBM * LDSS];
    __shared__ _Float16 sB[FBN * LDSS];

    const int tid  = threadIdx.x;
    const int lane = tid & 31;
    const int wave = tid >> 5;

    const long block_m = (long)blockIdx.y * FBM;
    const long block_n = (long)blockIdx.x * FBN;

    const int wm = (wave & 3) * 32;
    const int wn = (wave >> 2) * 64;

    v8f acc[2][4] = {};

    const int row0 = tid >> 4;
    const int c4   = (tid & 15) * 4;

    float4 ra[8];
    int4   rw[8];
    auto gload = [&](int k0) {
        #pragma unroll
        for (int j = 0; j < 8; ++j) {
            const int r = row0 + 16 * j;
            ra[j] = *(const float4*)(A  + (size_t)(block_m + r) * K_DIM + k0 + c4);
            rw[j] = *(const int4*)  (Wq + (size_t)(block_n + r) * K_DIM + k0 + c4);
        }
    };

    gload(0);

    for (int k0 = 0; k0 < K_DIM; k0 += FBK) {
        __syncthreads();
        #pragma unroll
        for (int j = 0; j < 8; ++j) {
            const int r = row0 + 16 * j;
            v4h ha, hb;
            ha[0] = (_Float16)ra[j].x;        ha[1] = (_Float16)ra[j].y;
            ha[2] = (_Float16)ra[j].z;        ha[3] = (_Float16)ra[j].w;
            hb[0] = (_Float16)(short)rw[j].x; hb[1] = (_Float16)(short)rw[j].y;
            hb[2] = (_Float16)(short)rw[j].z; hb[3] = (_Float16)(short)rw[j].w;
            *(v4h*)(sA + r * LDSS + c4) = ha;
            *(v4h*)(sB + r * LDSS + c4) = hb;
        }
        __syncthreads();

        if (k0 + FBK < K_DIM) gload(k0 + FBK);

        #pragma unroll
        for (int kk = 0; kk < FBK; kk += 32) {
            v16h af[2], bf[4];
            #pragma unroll
            for (int mi = 0; mi < 2; ++mi) af[mi] = frag_a(sA, wm + mi * 16, lane, kk);
            #pragma unroll
            for (int ni = 0; ni < 4; ++ni) bf[ni] = frag_b(sB, wn + ni * 16, lane, kk);
            #pragma unroll
            for (int mi = 0; mi < 2; ++mi)
                #pragma unroll
                for (int ni = 0; ni < 4; ++ni)
                    acc[mi][ni] = __builtin_amdgcn_wmma_f32_16x16x32_f16(
                        false, af[mi], false, bf[ni], (short)0, acc[mi][ni], false, false);
        }
    }

    const float s = scale[0];
    #pragma unroll
    for (int mi = 0; mi < 2; ++mi) {
        const long m0 = block_m + wm + mi * 16 + ((lane >> 4) << 3);
        #pragma unroll
        for (int ni = 0; ni < 4; ++ni) {
            const long n   = block_n + wn + ni * 16 + (lane & 15);
            const float bv = bias[n];
            #pragma unroll
            for (int r = 0; r < 8; ++r)
                Out[(m0 + r) * (long)N_DIM + n] = acc[mi][ni][r] * s + bv;
        }
    }
}

extern "C" void kernel_launch(void* const* d_in, const int* in_sizes, int n_in,
                              void* d_out, int out_size, void* d_ws, size_t ws_size,
                              hipStream_t stream) {
    const float* input   = (const float*)d_in[0];  // [8,512,4096] fp32
    const int*   qweight = (const int*)  d_in[1];  // [4096,4096] int32 (int8 range)
    const float* wscale  = (const float*)d_in[2];  // [1]
    const float* bias    = (const float*)d_in[3];  // [4096]
    float*       out     = (float*)d_out;          // [8,512,4096] fp32

    const size_t elems = (size_t)M_DIM * K_DIM;              // == N*K too
    const size_t need  = 2 * elems * sizeof(_Float16);       // 64 MB

    if (ws_size >= need) {
        _Float16* Ah = (_Float16*)d_ws;
        _Float16* Wh = Ah + elems;
        convert_a_kernel<<<(unsigned)(elems / 2048), 256, 0, stream>>>(input, Ah);
        convert_w_kernel<<<(unsigned)(elems / 2048), 256, 0, stream>>>(qweight, Wh);
        dim3 grid(N_DIM / GBN, M_DIM / GBM);                 // 16 x 32
        gemm_f16_wmma_kernel<<<grid, 256, 0, stream>>>(Ah, Wh, wscale, bias, out);
    } else {
        dim3 grid(N_DIM / FBN, M_DIM / FBM);                 // 32 x 32
        fused_int8_linear_kernel<<<grid, 256, 0, stream>>>(input, qweight, wscale, bias, out);
    }
}
